// EmbeddingLayer_8864812499460
// MI455X (gfx1250) — compile-verified, matrix operand order checked by reading, not run on hardware
//
#include <hip/hip_runtime.h>

// ---------------------------------------------------------------------------
// EmbeddingLayer on MI455X (gfx1250, wave32, WMMA 16x16x32 f16 -> f32 acc)
//
// out = relu( nf@W1 + (adj@prev)@W2 + Sp*u + Sn*v )
//
// Exact collapse of the (B,N,N,D) relu(e*W4) tensor:
//   relu(e*w) = max(w,0)*max(e,0) + min(w,0)*min(e,0)
//   Sp[b,i] = sum_j adj*max(e,0),  Sn[b,i] = sum_j adj*min(e,0)
//   u = relu(W4)@W3, v = min(W4,0)@W3  ->  x3 = Sp*u + Sn*v
//
// One block (128 thr = 4 waves) per 16-row strip -> 512 blocks / 2048 waves:
//   phase 1: block cooperatively stages the 16x64 adj tile (f32->f16) in LDS
//            (coalesced loads, Sp/Sn accumulated from the same values);
//            each wave WMMAs its own 16-col N-tile against pre-transposed
//            f16 prev. adj/ef read exactly once from HBM.
//   C-frag -> LDS (padded stride) -> A-frag transpose.
//   phase 2: out = relu(T@W2 + nf@W1 + Sp*u + Sn*v) via WMMA.
// ---------------------------------------------------------------------------

typedef __attribute__((ext_vector_type(16))) _Float16 v16h;
typedef __attribute__((ext_vector_type(8)))  _Float16 v8h;
typedef __attribute__((ext_vector_type(8)))  float    v8f;

#define B_  16
#define N_  512
#define D_  64
#define FN_ 32
#define TSTRIDE 72   // LDS row stride in halves (144 B) -> 16 distinct banks

static __device__ __forceinline__ v16h cat8(v8h lo, v8h hi) {
    return __builtin_shufflevector(lo, hi, 0,1,2,3,4,5,6,7,8,9,10,11,12,13,14,15);
}

static __device__ __forceinline__ v8f wmma16(v16h a, v16h b, v8f c) {
    // D = A(16x32 f16) x B(32x16 f16) + C(16x16 f32)
    return __builtin_amdgcn_wmma_f32_16x16x32_f16(
        /*neg_a=*/false, a, /*neg_b=*/false, b,
        /*c_mod=*/(short)0, c, /*reuse_a=*/false, /*reuse_b=*/false);
}

// ---------------------------------------------------------------------------
// Small precompute: W2T/W1T (f16, col-major so B-fragments are K-contiguous),
// u[d'] = sum_d relu(W4[d]) * W3[d,d'],  v[d'] = sum_d min(W4[d],0) * W3[d,d']
// ---------------------------------------------------------------------------
__global__ void k_prep_small(const float* __restrict__ W1, const float* __restrict__ W2,
                             const float* __restrict__ W3, const float* __restrict__ W4,
                             _Float16* __restrict__ W1T, _Float16* __restrict__ W2T,
                             float* __restrict__ u, float* __restrict__ v) {
    int tid = threadIdx.x;
    for (int idx = tid; idx < 64 * 64; idx += 256) {      // W2T[col*64+k] = W2[k*64+col]
        int col = idx >> 6, k = idx & 63;
        W2T[idx] = (_Float16)W2[k * 64 + col];
    }
    for (int idx = tid; idx < 64 * 32; idx += 256) {      // W1T[col*32+k] = W1[k*64+col]
        int col = idx >> 5, k = idx & 31;
        W1T[idx] = (_Float16)W1[k * 64 + col];
    }
    if (tid < 64) {
        float su = 0.f, sv = 0.f;
        for (int d = 0; d < 64; ++d) {
            float w  = W4[d];
            float w3 = W3[d * 64 + tid];
            su += fmaxf(w, 0.f) * w3;
            sv += fminf(w, 0.f) * w3;
        }
        u[tid] = su;
        v[tid] = sv;
    }
}

// ---------------------------------------------------------------------------
// Big conversions: PT[b][d][j] = (f16) prev[b][j][d]  (transposed, K-contiguous)
//                  NH[row][f]  = (f16) nf[row][f]
// ---------------------------------------------------------------------------
__global__ void k_prep_big(const float* __restrict__ prev, const float* __restrict__ nf,
                           _Float16* __restrict__ PT, _Float16* __restrict__ NH) {
    const int nPT = B_ * D_ * N_;           // 524288
    const int total = nPT + B_ * N_ * FN_;  // + 262144
    for (int idx = blockIdx.x * blockDim.x + threadIdx.x; idx < total;
         idx += gridDim.x * blockDim.x) {
        if (idx < nPT) {
            int b = idx / (D_ * N_);
            int d = (idx / N_) & (D_ - 1);
            int j = idx & (N_ - 1);
            PT[idx] = (_Float16)prev[((size_t)b * N_ + j) * D_ + d];
        } else {
            int t = idx - nPT;
            NH[t] = (_Float16)nf[t];
        }
    }
}

// ---------------------------------------------------------------------------
// Fused main kernel. One block (4 waves) per 16-row strip; wave w owns
// N-tile w. 512 blocks -> 2048 waves.
// ---------------------------------------------------------------------------
__global__ void __launch_bounds__(128)
k_main(const float* __restrict__ adj, const float* __restrict__ ef,
       const _Float16* __restrict__ PT, const _Float16* __restrict__ NH,
       const _Float16* __restrict__ W2T, const _Float16* __restrict__ W1T,
       const float* __restrict__ u, const float* __restrict__ v,
       float* __restrict__ out) {
    __shared__ __align__(16) _Float16 Al[16 * TSTRIDE];  // staged adj f16 (2304 B)
    __shared__ __align__(16) _Float16 Tl[16 * TSTRIDE];  // T tile          (2304 B)
    __shared__ float spl[16], snl[16];

    int tid  = threadIdx.x;
    int w    = tid >> 5;                         // wave = N-tile 0..3
    int b    = blockIdx.x >> 5;                  // batch
    int mt   = blockIdx.x & 31;                  // 16-row strip in batch
    int lane = tid & 31;
    int r    = lane & 15;
    int hi   = lane >> 4;

    // staging role: thread t covers row tr, cols kc..kc+7 of each 16x64 stage
    int tr = tid >> 3;                           // 0..15
    int kc = (tid & 7) * 8;                      // 0..56

    const float* Ar = adj + ((size_t)b * N_ + mt * 16 + tr) * N_;
    const float* Er = ef  + ((size_t)b * N_ + mt * 16 + tr) * N_;
    const _Float16* Bp = PT + ((size_t)b * D_ + w * 16 + r) * N_;

    v8f acc = v8f{};
    float sp = 0.f, sn = 0.f;

    // ---------------- phase 1: T = adj @ prev, fused Sp/Sn --------------
    for (int kb = 0; kb < N_; kb += 64) {
        // cooperative stage: 1024 adj + 1024 ef values, fully coalesced
        const float* ap = Ar + kb + kc;
        const float* ep = Er + kb + kc;
        float4 f0 = *(const float4*)(ap);
        float4 f1 = *(const float4*)(ap + 4);
        float4 e0 = *(const float4*)(ep);
        float4 e1 = *(const float4*)(ep + 4);

        sp += f0.x * fmaxf(e0.x, 0.f) + f0.y * fmaxf(e0.y, 0.f)
            + f0.z * fmaxf(e0.z, 0.f) + f0.w * fmaxf(e0.w, 0.f)
            + f1.x * fmaxf(e1.x, 0.f) + f1.y * fmaxf(e1.y, 0.f)
            + f1.z * fmaxf(e1.z, 0.f) + f1.w * fmaxf(e1.w, 0.f);
        sn += f0.x * fminf(e0.x, 0.f) + f0.y * fminf(e0.y, 0.f)
            + f0.z * fminf(e0.z, 0.f) + f0.w * fminf(e0.w, 0.f)
            + f1.x * fminf(e1.x, 0.f) + f1.y * fminf(e1.y, 0.f)
            + f1.z * fminf(e1.z, 0.f) + f1.w * fminf(e1.w, 0.f);

        v8h h;
        h[0] = (_Float16)f0.x; h[1] = (_Float16)f0.y;
        h[2] = (_Float16)f0.z; h[3] = (_Float16)f0.w;
        h[4] = (_Float16)f1.x; h[5] = (_Float16)f1.y;
        h[6] = (_Float16)f1.z; h[7] = (_Float16)f1.w;
        *(v8h*)&Al[tr * TSTRIDE + kc] = h;       // ds_store_b128
        __syncthreads();

        // two WMMA K-steps per stage; A-frags shared from LDS
        #pragma unroll
        for (int ks = 0; ks < 2; ++ks) {
            int kk = ks * 32;
            const _Float16* la = &Al[r * TSTRIDE + kk + hi * 8];
            v16h a = cat8(*(const v8h*)la, *(const v8h*)(la + 16));
            const _Float16* bp = Bp + kb + kk + hi * 8;
            v16h bb = cat8(*(const v8h*)bp, *(const v8h*)(bp + 16));
            acc = wmma16(a, bb, acc);
        }
        __syncthreads();
    }

    // Sp/Sn: reduce the 8 threads covering each row (lane groups of 8)
    sp += __shfl_xor(sp, 4, 32);
    sp += __shfl_xor(sp, 2, 32);
    sp += __shfl_xor(sp, 1, 32);
    sn += __shfl_xor(sn, 4, 32);
    sn += __shfl_xor(sn, 2, 32);
    sn += __shfl_xor(sn, 1, 32);
    if ((tid & 7) == 0) { spl[tr] = sp; snl[tr] = sn; }

    // C-frag (lane holds col r of tile w, rows vr+8*hi) -> LDS [row][col]
    #pragma unroll
    for (int vr = 0; vr < 8; ++vr) {
        Tl[(vr + 8 * hi) * TSTRIDE + w * 16 + r] = (_Float16)acc[vr];
    }
    __syncthreads();

    // ---------------- phase 2: out = relu(T@W2 + nf@W1 + Sp*u + Sn*v) ---
    v8f o = v8f{};

    // x2 = T @ W2 : A-frags from LDS (K-contiguous b128 loads)
    #pragma unroll
    for (int ks = 0; ks < 2; ++ks) {
        const _Float16* la = &Tl[r * TSTRIDE + ks * 32 + hi * 8];
        v16h a = cat8(*(const v8h*)la, *(const v8h*)(la + 16));
        const _Float16* bp = W2T + (w * 16 + r) * 64 + ks * 32 + hi * 8;
        v16h bb = cat8(*(const v8h*)bp, *(const v8h*)(bp + 16));
        o = wmma16(a, bb, o);
    }

    // x1 = nf @ W1 : one K=32 step
    {
        int gr = b * N_ + mt * 16 + r;
        const _Float16* ap = NH + (size_t)gr * FN_ + hi * 8;
        v16h a = cat8(*(const v8h*)ap, *(const v8h*)(ap + 16));
        const _Float16* bp = W1T + (w * 16 + r) * FN_ + hi * 8;
        v16h bb = cat8(*(const v8h*)bp, *(const v8h*)(bp + 16));
        o = wmma16(a, bb, o);
    }

    // epilogue: + Sp*u + Sn*v, relu, store
    int d  = w * 16 + r;
    float uu = u[d], vv = v[d];
    int rowbase = b * N_ + mt * 16 + 8 * hi;
    #pragma unroll
    for (int vr = 0; vr < 8; ++vr) {
        float s  = spl[8 * hi + vr];
        float n  = snl[8 * hi + vr];
        float vl = o[vr] + s * uu + n * vv;
        out[(size_t)(rowbase + vr) * D_ + d] = fmaxf(vl, 0.f);
    }
}

// ---------------------------------------------------------------------------
extern "C" void kernel_launch(void* const* d_in, const int* in_sizes, int n_in,
                              void* d_out, int out_size, void* d_ws, size_t ws_size,
                              hipStream_t stream) {
    (void)in_sizes; (void)n_in; (void)out_size; (void)ws_size;

    const float* prev = (const float*)d_in[0];  // (B,N,D)
    const float* adj  = (const float*)d_in[1];  // (B,N,N)
    const float* nf   = (const float*)d_in[2];  // (B,N,Fn)
    const float* ef   = (const float*)d_in[3];  // (B,N,N)
    const float* W1   = (const float*)d_in[4];  // (Fn,D)
    const float* W2   = (const float*)d_in[5];  // (D,D)
    const float* W3   = (const float*)d_in[6];  // (D,D)
    const float* W4   = (const float*)d_in[7];  // (1,D)
    float* out = (float*)d_out;

    char* ws = (char*)d_ws;
    _Float16* PT  = (_Float16*)(ws);             // 16*64*512 f16 = 1,048,576 B
    _Float16* NH  = (_Float16*)(ws + 1048576);   // 8192*32 f16   =   524,288 B
    _Float16* W2T = (_Float16*)(ws + 1572864);   // 64*64 f16     =     8,192 B
    _Float16* W1T = (_Float16*)(ws + 1581056);   // 64*32 f16     =     4,096 B
    float*    u   = (float*)(ws + 1585152);      // 64 f32
    float*    v   = (float*)(ws + 1585408);      // 64 f32

    k_prep_small<<<1, 256, 0, stream>>>(W1, W2, W3, W4, W1T, W2T, u, v);
    k_prep_big<<<512, 256, 0, stream>>>(prev, nf, PT, NH);
    k_main<<<512, 128, 0, stream>>>(adj, ef, PT, NH, W2T, W1T, u, v, out);
}
